// MPNN_76596446757056
// MI455X (gfx1250) — compile-verified, matrix operand order checked by reading, not run on hardware
//
#include <hip/hip_runtime.h>
#include <hip/hip_bf16.h>
#include <math.h>

// ---------------- problem constants ----------------
#define AGT   8
#define BSZC  16384
#define DH    128
#define INS   16
#define KDIM  42
#define NHEAD 3
#define ETILE 16
#define ROWS  128            // ETILE * AGT
#define NF    0.15430334996209191f   // 1/sqrt(42)

typedef _Float16 v16h __attribute__((ext_vector_type(16)));
typedef _Float16 h8v  __attribute__((ext_vector_type(8)));
typedef _Float16 h2v  __attribute__((ext_vector_type(2)));
typedef float    v8f  __attribute__((ext_vector_type(8)));

#define V8Z {0.f,0.f,0.f,0.f,0.f,0.f,0.f,0.f}

// ---------------- LDS layout (bytes) ----------------
#define SH 136                // h / msg tile stride (halfs): 128 + 8 pad
#define SQ 152                // qkv tile stride (halfs): 144 + 8 pad
#define SI 40                 // inp tile stride (halfs): 32 + 8 pad

#define L_HA   0
#define L_HB   (L_HA  + ROWS*SH*2)
#define L_MSG  (L_HB  + ROWS*SH*2)
#define L_QKV  (L_MSG + ROWS*SH*2)
#define L_MASK (L_QKV + ROWS*SQ*2)
#define L_TOTAL (L_MASK + ETILE*64)

// ---------------- workspace layout (halfs) ----------------
#define W_ENC 0                          // [128][32]   encW^T, K padded 16->32
#define W_QKV (W_ENC + 128*32)           // [3][144][128] per-head [Q|K|V]^T, KD padded 42->48
#define W_WO  (W_QKV + 3*144*128)        // [128][128]  Wo^T, K padded 126->128
#define W_UPD (W_WO  + 128*128)          // [128][256]  updW^T
#define W_GRU (W_UPD + 128*256)          // [2][128][256] rz combined + [128][128] WihnT + [128][128] WhhnT
#define W_V1  (W_GRU + 2*128*256 + 2*128*128)
#define W_POL (W_V1  + 128*128)

// ---------------- fragment helpers ----------------
// A/B fragment for v_wmma_f32_16x16x32_f16.
// ISA layout: lane l holds row (l&15); lane group g=(l>>4) selects K halves:
//   elements 0..7  -> K = g*8   .. g*8+7
//   elements 8..15 -> K = 16+g*8 .. 16+g*8+7
// p0 points at (row0, k0) of a row-major f16 tile with given stride (halfs).
static __device__ __forceinline__ v16h load_frag(const _Float16* p0, int stride, int lane) {
    const _Float16* p = p0 + (lane & 15) * stride + ((lane >> 4) << 3);
    union { v16h v; h8v h[2]; } u;
    u.h[0] = *(const h8v*)p;
    u.h[1] = *(const h8v*)(p + 16);
    return u.v;
}

// Same gather but source is row-major f32 (global), converted to f16.
static __device__ __forceinline__ v16h load_frag_f32(const float* p0, int stride, int lane) {
    const float* p = p0 + (lane & 15) * stride + ((lane >> 4) << 3);
    union { v16h v; _Float16 e[16]; } u;
#pragma unroll
    for (int i = 0; i < 8; ++i) u.e[i] = (_Float16)p[i];
#pragma unroll
    for (int i = 0; i < 8; ++i) u.e[8 + i] = (_Float16)p[16 + i];
    return u.v;
}

static __device__ __forceinline__ v8f wmma_f16(v16h a, v16h b, v8f c) {
    return __builtin_amdgcn_wmma_f32_16x16x32_f16(false, a, false, b, (short)0, c, false, false);
}

static __device__ __forceinline__ float sigmoidf_(float x) { return 1.f / (1.f + __expf(-x)); }

// ---------------- weight prep kernels ----------------
// dst[n*Kdst + k] = (k<Ksrc && n<Nsrc) ? src[k*Nsrc + n] : 0   (f32 -> f16 transpose+pad)
__global__ void k_transpose(const float* __restrict__ src, _Float16* __restrict__ dst,
                            int Ksrc, int Nsrc, int Kdst, int Ndst) {
    int idx = blockIdx.x * 256 + threadIdx.x;
    if (idx >= Kdst * Ndst) return;
    int n = idx / Kdst, k = idx % Kdst;
    float v = (k < Ksrc && n < Nsrc) ? src[k * Nsrc + n] : 0.f;
    dst[idx] = (_Float16)v;
}

// per-head [Q(48) | K(48) | V(48)] x 128, transposed, KD padded with zeros
__global__ void k_qkvT(const float* __restrict__ Wq, const float* __restrict__ Wk,
                       const float* __restrict__ Wv, _Float16* __restrict__ dst) {
    int idx = blockIdx.x * 256 + threadIdx.x;
    if (idx >= 3 * 144 * 128) return;
    int hh  = idx / (144 * 128);
    int rem = idx % (144 * 128);
    int j = rem / 128, k = rem % 128;
    int sel = j / 48, n = j % 48;
    const float* W = (sel == 0) ? Wq : (sel == 1) ? Wk : Wv;
    float v = (n < KDIM) ? W[(hh * 128 + k) * KDIM + n] : 0.f;
    dst[idx] = (_Float16)v;
}

// [gate r combined 128x256][gate z combined 128x256][WihnT 128x128][WhhnT 128x128]
__global__ void k_gruT(const float* __restrict__ Wih, const float* __restrict__ Whh,
                       _Float16* __restrict__ dst) {
    int idx = blockIdx.x * 256 + threadIdx.x;
    if (idx >= 2 * 128 * 256 + 2 * 128 * 128) return;
    float v;
    if (idx < 2 * 128 * 256) {
        int g = idx / (128 * 256);
        int rem = idx % (128 * 256);
        int n = rem / 256, k = rem % 256;
        v = (k < 128) ? Wih[k * 384 + g * 128 + n] : Whh[(k - 128) * 384 + g * 128 + n];
    } else {
        int i2 = idx - 2 * 128 * 256;
        int sel = i2 / (128 * 128);
        int rem = i2 % (128 * 128);
        int n = rem / 128, k = rem % 128;
        v = (sel == 0) ? Wih[k * 384 + 256 + n] : Whh[k * 384 + 256 + n];
    }
    dst[idx] = (_Float16)v;
}

// ---------------- main fused kernel ----------------
__global__ void __launch_bounds__(256)
mpnn_kernel(const float* __restrict__ inp, const float* __restrict__ state,
            const float* __restrict__ enc_b, const float* __restrict__ upd_b,
            const float* __restrict__ gru_bih, const float* __restrict__ gru_bhh,
            const float* __restrict__ val_b1, const float* __restrict__ val_w2,
            const float* __restrict__ val_b2, const float* __restrict__ pol_b,
            const _Float16* __restrict__ ws, float* __restrict__ out) {
    extern __shared__ unsigned char lds[];
    _Float16* hbuf0 = (_Float16*)(lds + L_HA);
    _Float16* hbuf1 = (_Float16*)(lds + L_HB);
    _Float16* msgb  = (_Float16*)(lds + L_MSG);
    _Float16* qkvb  = (_Float16*)(lds + L_QKV);
    _Float16* inpb  = (_Float16*)(lds + L_QKV);   // alias, only used in stage 0
    unsigned char* maskb = lds + L_MASK;

    const int tid  = threadIdx.x;
    const int lane = tid & 31;
    const int w    = tid >> 5;                    // wave id == M-tile id == agent id
    const int e0   = blockIdx.x * ETILE;

    // ---- stage 0: load inputs (f32 -> f16 tile, K padded to 32), build mask ----
    for (int i = tid; i < ROWS * 4; i += 256) {   // 4 float4 per row of 16
        int t = i >> 2, c4 = i & 3;
        int gr = ((t >> 4) << 14) + e0 + (t & 15);
        float4 f = ((const float4*)(inp + gr * INS))[c4];
        _Float16* d = inpb + t * SI + c4 * 4;
        d[0] = (_Float16)f.x; d[1] = (_Float16)f.y;
        d[2] = (_Float16)f.z; d[3] = (_Float16)f.w;
    }
    for (int i = tid; i < ROWS * 16; i += 256) {  // zero pad cols 16..31
        int t = i >> 4, c = i & 15;
        inpb[t * SI + 16 + c] = (_Float16)0.f;
    }
    for (int i = tid; i < ETILE * 64; i += 256) { // distance mask, diag never masked
        int e = i >> 6, q = (i >> 3) & 7, g = i & 7;
        const float* pq = inp + ((q << 14) + e0 + e) * INS + 2;
        const float* pg = inp + ((g << 14) + e0 + e) * INS + 2;
        float dx = pq[0] - pg[0], dy = pq[1] - pg[1];
        maskb[i] = ((dx * dx + dy * dy) > 1.0f && (q != g)) ? 1 : 0;
    }
    __syncthreads();

    // ---- encoder: h = relu(inp @ encW + b), M=128 K=32 N=128 ----
    {
        const _Float16* Bw = ws + W_ENC;
        v16h a = load_frag(inpb + (w * 16) * SI, SI, lane);
        for (int nt = 0; nt < 8; ++nt) {
            v16h b = load_frag(Bw + (nt * 16) * 32, 32, lane);
            v8f c = V8Z;
            c = wmma_f16(a, b, c);
            int n = lane & 15, g = lane >> 4;
            float bias = enc_b[nt * 16 + n];
            _Float16* drow = hbuf0 + (w * 16 + g * 8) * SH + nt * 16 + n;
#pragma unroll
            for (int r = 0; r < 8; ++r) {
                float v = c[r] + bias; v = v > 0.f ? v : 0.f;
                drow[r * SH] = (_Float16)v;
            }
        }
    }
    __syncthreads();

    _Float16* hcur = hbuf0;
    _Float16* hnxt = hbuf1;

    // ---- K_ROUNDS of message passing ----
    for (int round = 0; round < 3; ++round) {
        for (int hh = 0; hh < NHEAD; ++hh) {
            // QKV GEMM: M=128 K=128 N=144  -> qkvb (Q:0..47, K:48..95, V:96..143)
            const _Float16* Bw = ws + W_QKV + hh * 144 * 128;
            for (int nt = 0; nt < 9; ++nt) {
                v8f c = V8Z;
#pragma unroll
                for (int ks = 0; ks < 4; ++ks) {
                    v16h a = load_frag(hcur + (w * 16) * SH + ks * 32, SH, lane);
                    v16h b = load_frag(Bw + (nt * 16) * 128 + ks * 32, 128, lane);
                    c = wmma_f16(a, b, c);
                }
                int n = lane & 15, g = lane >> 4;
                _Float16* drow = qkvb + (w * 16 + g * 8) * SQ + nt * 16 + n;
#pragma unroll
                for (int r = 0; r < 8; ++r) drow[r * SQ] = (_Float16)c[r];
            }
            __syncthreads();

            // masked 8x8 attention per env (VALU; tiny fraction of FLOPs)
            if (lane < 16) {
                int e  = 2 * w + (lane >> 3);
                int q  = lane & 7;
                int tq = q * 16 + e;
                const _Float16* Qp = qkvb + tq * SQ;           // cols 0..41
                float s[8];
#pragma unroll
                for (int g = 0; g < 8; ++g) {
                    const _Float16* Kp = qkvb + (g * 16 + e) * SQ + 48;
                    float acc = 0.f;
                    for (int kk = 0; kk < 21; ++kk) {
                        h2v qv = *(const h2v*)(Qp + kk * 2);
                        h2v kv = *(const h2v*)(Kp + kk * 2);
                        acc += (float)qv[0] * (float)kv[0] + (float)qv[1] * (float)kv[1];
                    }
                    s[g] = maskb[e * 64 + q * 8 + g] ? -3.0e38f : acc * NF;
                }
                float mx = s[0];
#pragma unroll
                for (int g = 1; g < 8; ++g) mx = fmaxf(mx, s[g]);
                float p[8], sum = 0.f;
#pragma unroll
                for (int g = 0; g < 8; ++g) {
                    p[g] = (s[g] < -1.0e38f) ? 0.f : __expf(s[g] - mx);
                    sum += p[g];
                }
                float inv = 1.f / sum;
                _Float16* dp = hnxt + tq * SH + hh * KDIM;
                for (int v2 = 0; v2 < 21; ++v2) {
                    float a0 = 0.f, a1 = 0.f;
#pragma unroll
                    for (int g = 0; g < 8; ++g) {
                        h2v vv = *(const h2v*)(qkvb + (g * 16 + e) * SQ + 96 + v2 * 2);
                        float pw = p[g] * inv;
                        a0 += pw * (float)vv[0];
                        a1 += pw * (float)vv[1];
                    }
                    h2v o; o[0] = (_Float16)a0; o[1] = (_Float16)a1;
                    *(h2v*)(dp + v2 * 2) = o;
                }
                if (hh == 0) {  // zero padded cols 126..127 of heads buffer
                    h2v z; z[0] = (_Float16)0.f; z[1] = (_Float16)0.f;
                    *(h2v*)(hnxt + tq * SH + 126) = z;
                }
            }
            __syncthreads();
        }

        // msg = heads @ Wo : M=128 K=128 N=128
        {
            const _Float16* Bw = ws + W_WO;
            for (int nt = 0; nt < 8; ++nt) {
                v8f c = V8Z;
#pragma unroll
                for (int ks = 0; ks < 4; ++ks) {
                    v16h a = load_frag(hnxt + (w * 16) * SH + ks * 32, SH, lane);
                    v16h b = load_frag(Bw + (nt * 16) * 128 + ks * 32, 128, lane);
                    c = wmma_f16(a, b, c);
                }
                int n = lane & 15, g = lane >> 4;
                _Float16* drow = msgb + (w * 16 + g * 8) * SH + nt * 16 + n;
#pragma unroll
                for (int r = 0; r < 8; ++r) drow[r * SH] = (_Float16)c[r];
            }
        }
        __syncthreads();

        // h' = relu([h | msg] @ updW + b) : M=128 K=256 N=128
        {
            const _Float16* Bw = ws + W_UPD;
            for (int nt = 0; nt < 8; ++nt) {
                v8f c = V8Z;
#pragma unroll
                for (int ks = 0; ks < 8; ++ks) {
                    const _Float16* Ab = (ks < 4) ? (hcur + (w * 16) * SH + ks * 32)
                                                  : (msgb + (w * 16) * SH + (ks - 4) * 32);
                    v16h a = load_frag(Ab, SH, lane);
                    v16h b = load_frag(Bw + (nt * 16) * 256 + ks * 32, 256, lane);
                    c = wmma_f16(a, b, c);
                }
                int n = lane & 15, g = lane >> 4;
                float bias = upd_b[nt * 16 + n];
                _Float16* drow = hnxt + (w * 16 + g * 8) * SH + nt * 16 + n;
#pragma unroll
                for (int r = 0; r < 8; ++r) {
                    float v = c[r] + bias; v = v > 0.f ? v : 0.f;
                    drow[r * SH] = (_Float16)v;
                }
            }
        }
        __syncthreads();
        _Float16* tmp = hcur; hcur = hnxt; hnxt = tmp;
    }

    // ---- GRU cell: per (M-tile, N-tile) compute r,z,xn,hn with 4 WMMA chains ----
    {
        const _Float16* Brz = ws + W_GRU;                 // [2][128][256]
        const _Float16* Bxn = ws + W_GRU + 2 * 128 * 256; // [128][128]
        const _Float16* Bhn = Bxn + 128 * 128;            // [128][128]
        const int grbase = (w << 14) + e0;                // agent w, envs e0..e0+15
        const float* staterow = state + (size_t)grbase * DH;
        for (int nt = 0; nt < 8; ++nt) {
            v8f cr = V8Z, cz = V8Z, cxn = V8Z, chn = V8Z;
#pragma unroll
            for (int ks = 0; ks < 4; ++ks) {              // K 0..127 : h
                v16h a = load_frag(hcur + (w * 16) * SH + ks * 32, SH, lane);
                cr  = wmma_f16(a, load_frag(Brz + (nt * 16) * 256 + ks * 32, 256, lane), cr);
                cz  = wmma_f16(a, load_frag(Brz + 128 * 256 + (nt * 16) * 256 + ks * 32, 256, lane), cz);
                cxn = wmma_f16(a, load_frag(Bxn + (nt * 16) * 128 + ks * 32, 128, lane), cxn);
            }
#pragma unroll
            for (int ks = 0; ks < 4; ++ks) {              // K 128..255 : state (from global f32)
                v16h a = load_frag_f32(staterow + ks * 32, DH, lane);
                cr  = wmma_f16(a, load_frag(Brz + (nt * 16) * 256 + 128 + ks * 32, 256, lane), cr);
                cz  = wmma_f16(a, load_frag(Brz + 128 * 256 + (nt * 16) * 256 + 128 + ks * 32, 256, lane), cz);
                chn = wmma_f16(a, load_frag(Bhn + (nt * 16) * 128 + ks * 32, 128, lane), chn);
            }
            int n = lane & 15, g = lane >> 4;
            int col = nt * 16 + n;
            float br  = gru_bih[col]       + gru_bhh[col];
            float bz  = gru_bih[128 + col] + gru_bhh[128 + col];
            float bxn = gru_bih[256 + col];
            float bhn = gru_bhh[256 + col];
#pragma unroll
            for (int r = 0; r < 8; ++r) {
                int m = g * 8 + r;
                float sv = state[(size_t)(grbase + m) * DH + col];
                float rr = sigmoidf_(cr[r] + br);
                float zz = sigmoidf_(cz[r] + bz);
                float nn = tanhf(cxn[r] + bxn + rr * (chn[r] + bhn));
                float hv = (1.f - zz) * nn + zz * sv;
                out[(size_t)(grbase + m) * 257 + 129 + col] = hv;
                hnxt[(w * 16 + m) * SH + col] = (_Float16)hv;
            }
        }
    }
    __syncthreads();

    // ---- heads: value MLP layer1 -> qkvb scratch; policy -> out directly ----
    {
        const _Float16* Bw = ws + W_V1;
        for (int nt = 0; nt < 8; ++nt) {
            v8f c = V8Z;
#pragma unroll
            for (int ks = 0; ks < 4; ++ks) {
                v16h a = load_frag(hnxt + (w * 16) * SH + ks * 32, SH, lane);
                v16h b = load_frag(Bw + (nt * 16) * 128 + ks * 32, 128, lane);
                c = wmma_f16(a, b, c);
            }
            int n = lane & 15, g = lane >> 4;
            float bias = val_b1[nt * 16 + n];
            _Float16* drow = qkvb + (w * 16 + g * 8) * SQ + nt * 16 + n;
#pragma unroll
            for (int r = 0; r < 8; ++r) {
                float v = c[r] + bias; v = v > 0.f ? v : 0.f;
                drow[r * SQ] = (_Float16)v;
            }
        }
    }
    {
        const _Float16* Bw = ws + W_POL;
        const int grbase = (w << 14) + e0;
        for (int nt = 0; nt < 8; ++nt) {
            v8f c = V8Z;
#pragma unroll
            for (int ks = 0; ks < 4; ++ks) {
                v16h a = load_frag(hnxt + (w * 16) * SH + ks * 32, SH, lane);
                v16h b = load_frag(Bw + (nt * 16) * 128 + ks * 32, 128, lane);
                c = wmma_f16(a, b, c);
            }
            int n = lane & 15, g = lane >> 4;
            int col = nt * 16 + n;
            float bias = pol_b[col];
#pragma unroll
            for (int r = 0; r < 8; ++r) {
                int m = g * 8 + r;
                float v = c[r] + bias; v = v > 0.f ? v : 0.f;
                out[(size_t)(grbase + m) * 257 + 1 + col] = v;
            }
        }
    }
    __syncthreads();

    // ---- value layer2: 128-dot per row ----
    if (tid < ROWS) {
        int t = tid;
        int gr = ((t >> 4) << 14) + e0 + (t & 15);
        const _Float16* vp = qkvb + t * SQ;
        float acc = 0.f;
        for (int c2 = 0; c2 < 64; ++c2) {
            h2v v = *(const h2v*)(vp + c2 * 2);
            acc += (float)v[0] * val_w2[c2 * 2] + (float)v[1] * val_w2[c2 * 2 + 1];
        }
        out[(size_t)gr * 257] = acc + val_b2[0];
    }
}

// ---------------- host launch ----------------
extern "C" void kernel_launch(void* const* d_in, const int* in_sizes, int n_in,
                              void* d_out, int out_size, void* d_ws, size_t ws_size,
                              hipStream_t stream) {
    const float* inp     = (const float*)d_in[0];
    const float* state   = (const float*)d_in[1];
    const float* enc_W   = (const float*)d_in[2];
    const float* enc_b   = (const float*)d_in[3];
    const float* Wq      = (const float*)d_in[4];
    const float* Wk      = (const float*)d_in[5];
    const float* Wv      = (const float*)d_in[6];
    const float* Wo      = (const float*)d_in[7];
    const float* upd_W   = (const float*)d_in[8];
    const float* upd_b   = (const float*)d_in[9];
    const float* gru_Wih = (const float*)d_in[10];
    const float* gru_Whh = (const float*)d_in[11];
    const float* gru_bih = (const float*)d_in[12];
    const float* gru_bhh = (const float*)d_in[13];
    const float* val_W1  = (const float*)d_in[14];
    const float* val_b1  = (const float*)d_in[15];
    const float* val_W2  = (const float*)d_in[16];
    const float* val_b2  = (const float*)d_in[17];
    const float* pol_W   = (const float*)d_in[18];
    const float* pol_b   = (const float*)d_in[19];
    float* out = (float*)d_out;
    _Float16* ws = (_Float16*)d_ws;

    auto blocks = [](int n) { return (n + 255) / 256; };
    k_transpose<<<blocks(128 * 32), 256, 0, stream>>>(enc_W, ws + W_ENC, 16, 128, 32, 128);
    k_qkvT<<<blocks(3 * 144 * 128), 256, 0, stream>>>(Wq, Wk, Wv, ws + W_QKV);
    k_transpose<<<blocks(128 * 128), 256, 0, stream>>>(Wo, ws + W_WO, 126, 128, 128, 128);
    k_transpose<<<blocks(128 * 256), 256, 0, stream>>>(upd_W, ws + W_UPD, 256, 128, 256, 128);
    k_gruT<<<blocks(2 * 128 * 256 + 2 * 128 * 128), 256, 0, stream>>>(gru_Wih, gru_Whh, ws + W_GRU);
    k_transpose<<<blocks(128 * 128), 256, 0, stream>>>(val_W1, ws + W_V1, 128, 128, 128, 128);
    k_transpose<<<blocks(128 * 128), 256, 0, stream>>>(pol_W, ws + W_POL, 128, 128, 128, 128);

    mpnn_kernel<<<BSZC / ETILE, 256, L_TOTAL, stream>>>(
        inp, state, enc_b, upd_b, gru_bih, gru_bhh,
        val_b1, val_W2, val_b2, pol_b, ws, out);
}
